// MiniMaxedModule_14585708937680
// MI455X (gfx1250) — matrix operand order repaired and checked by reading, not compiled, not run on hardware
//
#include <hip/hip_runtime.h>
#include <hip/hip_bf16.h>

typedef __attribute__((ext_vector_type(16))) _Float16 v16h;
typedef __attribute__((ext_vector_type(8)))  _Float16 v8h;
typedef __attribute__((ext_vector_type(8)))  float    v8f;

#define BATCH  16
#define BRANCH 8
#define HID    256
#define DSTRIDE 260   // LDS row stride for D to avoid bank conflicts

__device__ __forceinline__ float fast_tanh(float x) {
#if __has_builtin(__builtin_amdgcn_tanhf)
    return __builtin_amdgcn_tanhf(x);           // v_tanh_f32 (CDNA5 TRANS op)
#else
    // tanh(x) = 1 - 2/(exp(2x)+1); exp via v_exp_f32 (2^y)
    float e = __builtin_exp2f(x * 2.8853900817779268f);
    return 1.0f - 2.0f * __builtin_amdgcn_rcpf(e + 1.0f);
#endif
}

// ---------------------------------------------------------------------------
// Kernel 1: A[16][256] = x@W1 + b1 ; D[8][256] = deltas@W1, via WMMA f16->f32.
// One wave. M = 32 rows (x:0-15, deltas:16-23, pad:24-31), K = 64, N = 256.
// 2 M-passes x 16 N-tiles x 2 K-steps of v_wmma_f32_16x16x32_f16.
// ---------------------------------------------------------------------------

// A fragment: lane holds row r; elements cover K = hi*16 + {0..15}
// (matches B fragment). Contiguous & 16B-aligned in LDS -> two b128 loads.
__device__ __forceinline__ v16h load_afrag(const _Float16* rowp, int ks, int hi) {
    const _Float16* p = rowp + ks * 32 + hi * 16;
    v8h lo = *(const v8h*)p;
    v8h hh = *(const v8h*)(p + 8);
    return __builtin_shufflevector(lo, hh, 0, 1, 2, 3, 4, 5, 6, 7,
                                           8, 9, 10, 11, 12, 13, 14, 15);
}

// B fragment: lane holds col n; halves cover K = hi*16 + i (transpose gather).
__device__ __forceinline__ v16h load_bfrag(const _Float16 (*Bh)[256],
                                           int nt, int ks, int n, int hi) {
    v16h bf;
    #pragma unroll
    for (int i = 0; i < 16; ++i)
        bf[i] = Bh[ks * 32 + hi * 16 + i][nt * 16 + n];
    return bf;
}

__global__ __launch_bounds__(32)
void precompute_AD_wmma(const float* __restrict__ x,
                        const float* __restrict__ deltas,
                        const float* __restrict__ W1,
                        const float* __restrict__ b1,
                        float* __restrict__ Aout,   // [16*256]
                        float* __restrict__ Dout) { // [8*256]
    __shared__ _Float16 Ah[32][64];    // [Mrow][K]
    __shared__ _Float16 Bh[64][256];   // [K][N]

    const int lane = threadIdx.x;

    // Stage inputs into LDS as f16
    for (int idx = lane; idx < 32 * 64; idx += 32) {
        int r = idx >> 6, k = idx & 63;
        float v = (r < 16) ? x[r * 64 + k]
                : (r < 24) ? deltas[(r - 16) * 64 + k]
                : 0.0f;
        Ah[r][k] = (_Float16)v;
    }
    for (int idx = lane; idx < 64 * 256; idx += 32) {
        Bh[idx >> 8][idx & 255] = (_Float16)W1[idx];
    }
    __syncthreads();

    const int r  = lane & 15;   // M row within tile (A) / N col (B, C/D)
    const int hi = lane >> 4;   // half-wave selector

    // ---- M-pass 0: rows 0-15 (x) -> Aout = x@W1 + b1 ----
    for (int nt = 0; nt < 16; ++nt) {
        v8f cacc = {};
        #pragma unroll
        for (int ks = 0; ks < 2; ++ks) {
            v16h a  = load_afrag(&Ah[r][0], ks, hi);
            v16h bf = load_bfrag(Bh, nt, ks, r, hi);
            cacc = __builtin_amdgcn_wmma_f32_16x16x32_f16(
                       false, a, false, bf, (short)0, cacc, false, false);
        }
        // C/D layout: VGPR v -> M = hi*8 + v, col = nt*16 + r
        const int col = nt * 16 + r;
        const float bias = b1[col];
        #pragma unroll
        for (int v = 0; v < 8; ++v)
            Aout[(hi * 8 + v) * 256 + col] = cacc[v] + bias;
    }

    // ---- M-pass 1: rows 16-31 (deltas + pad) -> Dout = deltas@W1 ----
    for (int nt = 0; nt < 16; ++nt) {
        v8f cacc = {};
        #pragma unroll
        for (int ks = 0; ks < 2; ++ks) {
            v16h a  = load_afrag(&Ah[16 + r][0], ks, hi);
            v16h bf = load_bfrag(Bh, nt, ks, r, hi);
            cacc = __builtin_amdgcn_wmma_f32_16x16x32_f16(
                       false, a, false, bf, (short)0, cacc, false, false);
        }
        const int col = nt * 16 + r;
        if (hi == 0) {   // rows 16-23 = D rows 0-7; rows 24-31 are pad
            #pragma unroll
            for (int v = 0; v < 8; ++v)
                Dout[v * 256 + col] = cacc[v];
        }
    }
}

// ---------------------------------------------------------------------------
// Kernel 2: per (b,m1,m2) block: evaluate all 512 descendant leaves and
// reduce m5(max) -> m4(min) -> m3(max). 8 waves = m3; lane = child(m5)*4+g.
// ---------------------------------------------------------------------------
__global__ __launch_bounds__(256)
void leaf_eval_minimax(const float* __restrict__ A,     // [16*256]
                       const float* __restrict__ D,     // [8*256]
                       const float* __restrict__ W2,    // [256]
                       float* __restrict__ nodeval) {   // [1024] = (b,m1,m2)
    __shared__ float s2[HID];               // A[b]+D[m1]+D[m2]
    __shared__ float Dl[8 * DSTRIDE];       // padded D rows
    __shared__ float w2l[HID];
    __shared__ float wred[8];

    const int tid = threadIdx.x;
    const int bid = blockIdx.x;
    const int b  = bid >> 6;
    const int m1 = (bid >> 3) & 7;
    const int m2 = bid & 7;

    // Load D (padded) and w2
    for (int idx = tid; idx < 8 * HID; idx += 256) {
        int m = idx >> 8, j = idx & 255;
        Dl[m * DSTRIDE + j] = D[idx];
    }
    w2l[tid] = W2[tid];
    __syncthreads();

    s2[tid] = A[b * 256 + tid] + Dl[m1 * DSTRIDE + tid] + Dl[m2 * DSTRIDE + tid];
    __syncthreads();

    const int m3   = tid >> 5;       // wave id
    const int lane = tid & 31;
    const int c    = lane >> 2;      // child = m5
    const int g    = lane & 3;       // dim group; lane covers j = 4*jj + g

    // Hoist S2 + D[m3] + D[c] into registers (64 VGPRs/lane).
    // D[c] is lane-invariant across the whole m4 loop, so fold it in here:
    // inner loop then needs only 2 LDS reads per element (D[m4], w2).
    float s23c[64];
    #pragma unroll
    for (int jj = 0; jj < 64; ++jj) {
        int j = 4 * jj + g;
        s23c[jj] = s2[j] + Dl[m3 * DSTRIDE + j] + Dl[c * DSTRIDE + j];
    }

    float vmin = 1.0e30f;
    for (int m4 = 0; m4 < 8; ++m4) {
        const float* D4 = &Dl[m4 * DSTRIDE + g];
        float acc = 0.0f;
        #pragma unroll
        for (int jj = 0; jj < 64; ++jj) {
            int j4 = 4 * jj;
            float xv = s23c[jj] + D4[j4];
            acc = fmaf(fast_tanh(xv), w2l[j4 + g], acc);
        }
        // sum partial dot across the 4 dim-group lanes of this child
        acc += __shfl_xor(acc, 1, 32);
        acc += __shfl_xor(acc, 2, 32);
        // leaf level (m5): max across the 8 children
        float vc = acc;
        vc = fmaxf(vc, __shfl_xor(vc, 4, 32));
        vc = fmaxf(vc, __shfl_xor(vc, 8, 32));
        vc = fmaxf(vc, __shfl_xor(vc, 16, 32));
        // m4 level: min
        vmin = fminf(vmin, vc);
    }

    if (lane == 0) wred[m3] = vmin;
    __syncthreads();
    if (tid == 0) {
        // m3 level: max across waves
        float vmax = wred[0];
        #pragma unroll
        for (int w = 1; w < 8; ++w) vmax = fmaxf(vmax, wred[w]);
        nodeval[bid] = vmax;
    }
}

// ---------------------------------------------------------------------------
// Kernel 3: m2(min) -> m1(max), add b2, write out[16].
// ---------------------------------------------------------------------------
__global__ __launch_bounds__(128)
void final_reduce(const float* __restrict__ nodeval,
                  const float* __restrict__ b2,
                  float* __restrict__ out) {
    const int t  = threadIdx.x;     // 0..127
    const int b  = t >> 3;
    const int m1 = t & 7;
    float v = 1.0e30f;
    #pragma unroll
    for (int m2 = 0; m2 < 8; ++m2)
        v = fminf(v, nodeval[b * 64 + m1 * 8 + m2]);   // m2 level: min
    // m1 level: max across the 8 lanes of this b
    v = fmaxf(v, __shfl_xor(v, 1, 32));
    v = fmaxf(v, __shfl_xor(v, 2, 32));
    v = fmaxf(v, __shfl_xor(v, 4, 32));
    if (m1 == 0) out[b] = v + b2[0];
}

extern "C" void kernel_launch(void* const* d_in, const int* in_sizes, int n_in,
                              void* d_out, int out_size, void* d_ws, size_t ws_size,
                              hipStream_t stream) {
    const float* x      = (const float*)d_in[0];  // [16*64]
    const float* deltas = (const float*)d_in[1];  // [8*64]
    const float* W1     = (const float*)d_in[2];  // [64*256]
    const float* b1     = (const float*)d_in[3];  // [256]
    const float* W2     = (const float*)d_in[4];  // [256]
    const float* b2     = (const float*)d_in[5];  // [1]
    float* out = (float*)d_out;                   // [16]

    float* wsf = (float*)d_ws;
    float* A  = wsf;            // 16*256 = 4096
    float* D  = wsf + 4096;     // 8*256  = 2048
    float* NV = wsf + 6144;     // 1024

    precompute_AD_wmma<<<1, 32, 0, stream>>>(x, deltas, W1, b1, A, D);
    leaf_eval_minimax<<<BATCH * BRANCH * BRANCH, 256, 0, stream>>>(A, D, W2, NV);
    final_reduce<<<1, 128, 0, stream>>>(NV, b2, out);
}
